// IntraViewDiffusion_45698452030226
// MI455X (gfx1250) — compile-verified, matrix operand order checked by reading, not run on hardware
//
#include <hip/hip_runtime.h>
#include <stdint.h>

typedef __attribute__((ext_vector_type(16))) _Float16 v16h;
typedef __attribute__((ext_vector_type(8)))  _Float16 v8h;
typedef __attribute__((ext_vector_type(8)))  float    v8f;

#define NV    3
#define NN    10000
#define DIN   256
#define DOUT  64
#define NPAD  10016      // NN rounded up to multiple of 32
#define NSTRIPS 625      // NN / 16
#define NITER 313        // NPAD / 32 key steps
#define BN_EPS 1e-5f

// Fragment helper: 16-bit A/B fragment per ISA layout.
// Lane half h = lane>>4. Element j<8 -> K = h*8+j ; j>=8 -> K = 16+h*8+(j-8).
// Both K ranges are 8 contiguous f16 -> two 16-byte vector loads (global or ds).
__device__ __forceinline__ v16h load_frag(const _Float16* p, int h) {
    v8h lo = *(const v8h*)(p + h * 8);
    v8h hi = *(const v8h*)(p + 16 + h * 8);
    v16h f;
#pragma unroll
    for (int i = 0; i < 8; ++i) { f[i] = lo[i]; f[i + 8] = hi[i]; }
    return f;
}

__device__ __forceinline__ v8f wmma_f16(v16h a, v16h b, v8f c) {
    return __builtin_amdgcn_wmma_f32_16x16x32_f16(false, a, false, b, (short)0, c, false, false);
}

// Async global->LDS 128-bit copy (gfx1250). VDST VGPR carries the raw LDS byte
// offset; per the FLAT aperture rules the low 32 bits of a generic LDS pointer
// ARE that offset. Tracked by ASYNCcnt.
__device__ __forceinline__ void async_copy_b128(void* lds_ptr, const void* gptr) {
    uint32_t lds_off = (uint32_t)(uintptr_t)lds_ptr;
    asm volatile("global_load_async_to_lds_b128 %0, %1, off"
                 :: "v"(lds_off), "v"(gptr) : "memory");
}
__device__ __forceinline__ void wait_async0() {
    asm volatile("s_wait_asynccnt 0x0" ::: "memory");
}

// ---------------- 1) convert x -> f16 ----------------
__global__ void cvt_x_kernel(const float* __restrict__ x, _Float16* __restrict__ xh, int total) {
    int i = blockIdx.x * blockDim.x + threadIdx.x;
    if (i < total) xh[i] = (_Float16)x[i];
}

// ---------------- 2) convert W -> f16, transposed [ (proj*V+v)*64 + c ][ k ] ----------------
__global__ void cvt_w_kernel(const float* __restrict__ Wq, const float* __restrict__ Wk,
                             const float* __restrict__ Wv, _Float16* __restrict__ Wt) {
    int i = blockIdx.x * blockDim.x + threadIdx.x;
    const int per = NV * DIN * DOUT;
    if (i >= 3 * per) return;
    int proj = i / per, rem = i % per;
    int v = rem / (DIN * DOUT); int r2 = rem % (DIN * DOUT);
    int k = r2 / DOUT, c = r2 % DOUT;
    const float* W = proj == 0 ? Wq : proj == 1 ? Wk : Wv;
    Wt[((size_t)(proj * NV + v) * DOUT + c) * DIN + k] = (_Float16)W[((size_t)v * DIN + k) * DOUT + c];
}

// ---------------- 3) projections via WMMA: h = x @ W + b (f32 out) ----------------
__global__ void proj_kernel(const _Float16* __restrict__ xh, const _Float16* __restrict__ Wt,
                            const float* __restrict__ bq, const float* __restrict__ bk,
                            const float* __restrict__ bv, float* __restrict__ hbuf) {
    const int wave = threadIdx.x >> 5;
    const int lane = threadIdx.x & 31;
    const int h = lane >> 4, r = lane & 15;
    const int strip = blockIdx.x * 8 + wave;
    const int v = blockIdx.y;
    const int proj = blockIdx.z;
    if (strip >= NSTRIPS) return;              // wave-uniform exit (no barriers here)
    const float* bias = proj == 0 ? bq : proj == 1 ? bk : bv;
    const _Float16* xrow = xh + ((size_t)v * NN + strip * 16 + r) * DIN;       // A: lane = row
    const _Float16* wt   = Wt + (size_t)(proj * NV + v) * DOUT * DIN;          // B: lane = col
    v8f acc[4] = {};
    for (int kc = 0; kc < DIN; kc += 32) {
        v16h a = load_frag(xrow + kc, h);
#pragma unroll
        for (int t = 0; t < 4; ++t) {
            v16h b = load_frag(wt + (size_t)(t * 16 + r) * DIN + kc, h);
            acc[t] = wmma_f16(a, b, acc[t]);
        }
    }
    float* hout = hbuf + (size_t)(proj * NV + v) * NN * DOUT;
#pragma unroll
    for (int t = 0; t < 4; ++t) {
        int col = t * 16 + r;
        float bb = bias[v * DOUT + col];
#pragma unroll
        for (int rr = 0; rr < 8; ++rr) {
            int row = strip * 16 + rr + 8 * h;   // C layout: M = rr + 8*half
            hout[(size_t)row * DOUT + col] = acc[t][rr] + bb;
        }
    }
}

// ---------------- 4) BN stats (deterministic 2-level reduction) + fold gamma/beta ----------------
__global__ void bn_stats_kernel(const float* __restrict__ hbuf,
                                const float* __restrict__ gq, const float* __restrict__ betaq,
                                const float* __restrict__ gk, const float* __restrict__ betak,
                                const float* __restrict__ gv, const float* __restrict__ betav,
                                float* __restrict__ ss) {
    const int pv = blockIdx.x;                 // proj*V + v
    const int proj = pv / NV, v = pv % NV;
    const int c = threadIdx.x & 63, part = threadIdx.x >> 6;
    const float* hp = hbuf + (size_t)pv * NN * DOUT;
    float s = 0.f, s2 = 0.f;
    for (int n = part; n < NN; n += 4) {
        float x = hp[(size_t)n * DOUT + c];
        s += x; s2 += x * x;
    }
    __shared__ float sh[2][4][DOUT];
    sh[0][part][c] = s; sh[1][part][c] = s2;
    __syncthreads();
    if (part == 0) {
        float sum = sh[0][0][c] + sh[0][1][c] + sh[0][2][c] + sh[0][3][c];
        float sq  = sh[1][0][c] + sh[1][1][c] + sh[1][2][c] + sh[1][3][c];
        float mean = sum / (float)NN;
        float var  = sq / (float)NN - mean * mean;     // biased, matches reference
        const float* g  = proj == 0 ? gq : proj == 1 ? gk : gv;
        const float* be = proj == 0 ? betaq : proj == 1 ? betak : betav;
        float scale = g[v * DOUT + c] * rsqrtf(var + BN_EPS);
        float shift = be[v * DOUT + c] - mean * scale;
        ss[pv * 2 * DOUT + c] = scale;
        ss[pv * 2 * DOUT + DOUT + c] = shift;
    }
}

// ---------------- 5) apply BN -> f16 q/k (row major, padded) + v transposed ----------------
__global__ void bn_apply_kernel(const float* __restrict__ hbuf, const float* __restrict__ ss,
                                _Float16* __restrict__ hh, _Float16* __restrict__ vt) {
    int i = blockIdx.x * blockDim.x + threadIdx.x;
    const int total = 9 * NN * DOUT;
    if (i >= total) return;
    int pv  = i / (NN * DOUT);
    int rem = i - pv * (NN * DOUT);
    int n = rem / DOUT, c = rem - n * DOUT;
    float val = hbuf[i] * ss[pv * 2 * DOUT + c] + ss[pv * 2 * DOUT + DOUT + c];
    _Float16 hv = (_Float16)val;
    hh[((size_t)pv * NPAD + n) * DOUT + c] = hv;
    if (pv >= 2 * NV) {                         // v projection also stored transposed
        int v = pv - 2 * NV;
        vt[((size_t)v * DOUT + c) * NPAD + n] = hv;
    }
}

// ---------------- 6) zero padded tail rows/cols ----------------
__global__ void pad_zero_kernel(_Float16* __restrict__ hh, _Float16* __restrict__ vt) {
    int i = blockIdx.x * blockDim.x + threadIdx.x;
    const int padRows = NPAD - NN;                       // 16
    const int totalHH = 9 * padRows * DOUT;              // 9216
    const int totalVT = NV * DOUT * padRows;             // 3072
    if (i < totalHH) {
        int pv = i / (padRows * DOUT);
        int rem = i - pv * (padRows * DOUT);
        int n = NN + rem / DOUT, c = rem % DOUT;
        hh[((size_t)pv * NPAD + n) * DOUT + c] = (_Float16)0.f;
    } else if (i < totalHH + totalVT) {
        int j = i - totalHH;
        int vc = j / padRows;
        int n  = NN + j % padRows;
        vt[(size_t)vc * NPAD + n] = (_Float16)0.f;
    }
}

// ---------------- 7) flash sigmoid-kernel attention ----------------
// 8 waves/block share double-buffered LDS k/v tiles filled by async copies
// (global_load_async_to_lds_b128, ASYNCcnt). Per 32-key step: 4 WMMAs for
// T = k.q^T (C layout == A layout for the next GEMM; lane = query row),
// per-lane sigmoid+mask, then 5 WMMAs: 4x O += sigma(S).v + rowsum via ones-B.
__global__ void __launch_bounds__(256) attn_kernel(const _Float16* __restrict__ hh,
                                                   const _Float16* __restrict__ vt,
                                                   float* __restrict__ out) {
    const int tid  = threadIdx.x;
    const int wave = tid >> 5;
    const int lane = tid & 31;
    const int h = lane >> 4, r = lane & 15;
    int strip = blockIdx.x * 8 + wave;
    if (strip >= NSTRIPS) strip = NSTRIPS - 1;   // clamp: all waves stay for barriers
    const int v = blockIdx.y;
    const _Float16* qh  = hh + (size_t)(0 * NV + v) * NPAD * DOUT;
    const _Float16* kh  = hh + (size_t)(1 * NV + v) * NPAD * DOUT;
    const _Float16* vtp = vt + (size_t)v * DOUT * NPAD;

    // Double-buffered tiles: k [32 keys][64 d], v^T [64 cols][32 keys]
    __shared__ __align__(16) _Float16 sk[2][32 * 64];
    __shared__ __align__(16) _Float16 sv[2][64 * 32];

    // Per-thread staging slots (one async b128 per tile per thread)
    const int krow = tid >> 3, kchunk = tid & 7;   // 32 rows x 8 chunks of 16B
    const int vcol = tid >> 2, vchunk = tid & 3;   // 64 cols x 4 chunks of 16B

    const _Float16* qrow = qh + (size_t)(strip * 16 + r) * DOUT;  // B-frag: lane = q row
    v16h qf0 = load_frag(qrow, h);
    v16h qf1 = load_frag(qrow + 32, h);
    v16h ones;
#pragma unroll
    for (int i = 0; i < 16; ++i) ones[i] = (_Float16)1.0f;

    // Prologue: stage tile 0
    async_copy_b128(&sk[0][krow * 64 + kchunk * 8], kh + (size_t)krow * DOUT + kchunk * 8);
    async_copy_b128(&sv[0][vcol * 32 + vchunk * 8], vtp + (size_t)vcol * NPAD + vchunk * 8);
    wait_async0();
    __syncthreads();

    v8f accO0 = {}, accO1 = {}, accO2 = {}, accO3 = {}, accR = {};
    int cur = 0;
    for (int it = 0; it < NITER; ++it) {
        const int kb = it * 32;
        if (it + 1 < NITER) {  // prefetch next tile into the other buffer
            const int nb = kb + 32;
            async_copy_b128(&sk[cur ^ 1][krow * 64 + kchunk * 8],
                            kh + (size_t)(nb + krow) * DOUT + kchunk * 8);
            async_copy_b128(&sv[cur ^ 1][vcol * 32 + vchunk * 8],
                            vtp + (size_t)vcol * NPAD + nb + vchunk * 8);
        }
        const _Float16* skb = &sk[cur][0];
        const _Float16* svb = &sv[cur][0];
        // T tiles from LDS (k A-frag: lane = key row; q B-frag resident)
        v8f t0 = {}, t1 = {};
        t0 = wmma_f16(load_frag(skb + (size_t)r * DOUT, h),             qf0, t0);
        t0 = wmma_f16(load_frag(skb + (size_t)r * DOUT + 32, h),        qf1, t0);
        t1 = wmma_f16(load_frag(skb + (size_t)(16 + r) * DOUT, h),      qf0, t1);
        t1 = wmma_f16(load_frag(skb + (size_t)(16 + r) * DOUT + 32, h), qf1, t1);
        // sigmoid + key mask, packed per-lane straight into the A fragment
        v16h s;
#pragma unroll
        for (int j = 0; j < 8; ++j) {
            int key0 = kb + 8 * h + j;
            float x0 = t0[j];
            s[j]     = (key0 < NN) ? (_Float16)(1.0f / (1.0f + __expf(-x0))) : (_Float16)0.f;
            int key1 = kb + 16 + 8 * h + j;
            float x1 = t1[j];
            s[j + 8] = (key1 < NN) ? (_Float16)(1.0f / (1.0f + __expf(-x1))) : (_Float16)0.f;
        }
        accO0 = wmma_f16(s, load_frag(svb + (size_t)(0 * 16 + r) * 32, h), accO0);
        accO1 = wmma_f16(s, load_frag(svb + (size_t)(1 * 16 + r) * 32, h), accO1);
        accO2 = wmma_f16(s, load_frag(svb + (size_t)(2 * 16 + r) * 32, h), accO2);
        accO3 = wmma_f16(s, load_frag(svb + (size_t)(3 * 16 + r) * 32, h), accO3);
        accR  = wmma_f16(s, ones, accR);   // row sums, same C layout as O
        wait_async0();       // next buffer landed
        __syncthreads();     // everyone done reading current buffer
        cur ^= 1;
    }
    float* op = out + (size_t)v * NN * DOUT;
#pragma unroll
    for (int rr = 0; rr < 8; ++rr) {
        int row = strip * 16 + rr + 8 * h;
        float inv = 1.0f / (accR[rr] + 1e-8f);
        op[(size_t)row * DOUT +  0 + r] = accO0[rr] * inv;
        op[(size_t)row * DOUT + 16 + r] = accO1[rr] * inv;
        op[(size_t)row * DOUT + 32 + r] = accO2[rr] * inv;
        op[(size_t)row * DOUT + 48 + r] = accO3[rr] * inv;
    }
}

extern "C" void kernel_launch(void* const* d_in, const int* in_sizes, int n_in,
                              void* d_out, int out_size, void* d_ws, size_t ws_size,
                              hipStream_t stream) {
    (void)in_sizes; (void)n_in; (void)out_size; (void)ws_size;
    const float* x     = (const float*)d_in[0];
    const float* Wq    = (const float*)d_in[1];
    const float* bq    = (const float*)d_in[2];
    const float* gq    = (const float*)d_in[3];
    const float* betaq = (const float*)d_in[4];
    const float* Wk    = (const float*)d_in[5];
    const float* bk    = (const float*)d_in[6];
    const float* gk    = (const float*)d_in[7];
    const float* betak = (const float*)d_in[8];
    const float* Wv    = (const float*)d_in[9];
    const float* bv    = (const float*)d_in[10];
    const float* gv    = (const float*)d_in[11];
    const float* betav = (const float*)d_in[12];
    float* out = (float*)d_out;

    char* ws = (char*)d_ws;
    size_t off = 0;
    auto alloc = [&](size_t bytes) {
        void* p = ws + off;
        off = (off + bytes + 255) & ~(size_t)255;
        return p;
    };
    float*    hbuf = (float*)   alloc((size_t)9 * NN * DOUT * 4);        // ~23 MB
    _Float16* hh   = (_Float16*)alloc((size_t)9 * NPAD * DOUT * 2);      // ~11.5 MB
    _Float16* vt   = (_Float16*)alloc((size_t)NV * DOUT * NPAD * 2);     // ~3.8 MB
    float*    ssb  = (float*)   alloc((size_t)9 * 2 * DOUT * 4);
    _Float16* xh   = (_Float16*)alloc((size_t)NV * NN * DIN * 2);        // ~15.4 MB
    _Float16* Wt   = (_Float16*)alloc((size_t)9 * DOUT * DIN * 2);

    { int total = NV * NN * DIN;
      cvt_x_kernel<<<(total + 255) / 256, 256, 0, stream>>>(x, xh, total); }
    { int total = 3 * NV * DIN * DOUT;
      cvt_w_kernel<<<(total + 255) / 256, 256, 0, stream>>>(Wq, Wk, Wv, Wt); }
    { dim3 grid((NSTRIPS + 7) / 8, NV, 3);
      proj_kernel<<<grid, 256, 0, stream>>>(xh, Wt, bq, bk, bv, hbuf); }
    bn_stats_kernel<<<9, 256, 0, stream>>>(hbuf, gq, betaq, gk, betak, gv, betav, ssb);
    { int total = 9 * NN * DOUT;
      bn_apply_kernel<<<(total + 255) / 256, 256, 0, stream>>>(hbuf, ssb, hh, vt); }
    pad_zero_kernel<<<48, 256, 0, stream>>>(hh, vt);
    { dim3 grid((NSTRIPS + 7) / 8, NV);
      attn_kernel<<<grid, 256, 0, stream>>>(hh, vt, out); }
}
